// NECBOW_56951266345326
// MI455X (gfx1250) — matrix operand order repaired and checked by reading, not compiled, not run on hardware
//
#include <hip/hip_runtime.h>
#include <hip/hip_bf16.h>
#include <stdint.h>

typedef __attribute__((ext_vector_type(16))) _Float16 v16h;
typedef __attribute__((ext_vector_type(8)))  float    v8f;

#define NS_CONST 10
#define TWO_W 8
#define DIM 256
#define CTX_DIM 2048   // TWO_W * DIM
#define KSTEPS 64      // CTX_DIM / 32

// ---------------------------------------------------------------------------
// Kernel 1: inclusive CDF scan of sampling weights (single block, chunked
// Hillis-Steele in LDS). V=50000 -> ~196 chunks of 256; negligible runtime.
// ---------------------------------------------------------------------------
__global__ __launch_bounds__(256) void necbow_cdf_scan(const float* __restrict__ w,
                                                       float* __restrict__ cdf, int V) {
    __shared__ float buf[256];
    const int tid = threadIdx.x;
    float running = 0.f;
    for (int base = 0; base < V; base += 256) {
        int i = base + tid;
        float x = (i < V) ? w[i] : 0.f;
        buf[tid] = x;
        __syncthreads();
        for (int off = 1; off < 256; off <<= 1) {
            float t = (tid >= off) ? buf[tid - off] : 0.f;
            __syncthreads();
            buf[tid] += t;
            __syncthreads();
        }
        if (i < V) cdf[i] = running + buf[tid];
        running += buf[255];
        __syncthreads();
    }
}

// ---------------------------------------------------------------------------
// Kernel 2: draw B*NS categorical samples via inverse-CDF binary search,
// uniforms from a PCG hash of the flat sample index (deterministic).
// ---------------------------------------------------------------------------
__global__ __launch_bounds__(256) void necbow_sample(const float* __restrict__ cdf,
                                                     int* __restrict__ noises,
                                                     int n, int V) {
    int idx = blockIdx.x * blockDim.x + threadIdx.x;
    if (idx >= n) return;
    uint32_t s = (uint32_t)idx * 747796405u + 2891336453u;
    s = ((s >> ((s >> 28) + 4u)) ^ s) * 277803737u;
    s = (s >> 22) ^ s;
    float u = (float)(s >> 8) * (1.0f / 16777216.0f);   // [0,1)
    float target = u * cdf[V - 1];
    int lo = 0, hi = V - 1;
    while (lo < hi) {
        int mid = (lo + hi) >> 1;
        if (cdf[mid] < target) lo = mid + 1; else hi = mid;
    }
    noises[idx] = lo;
}

// ---------------------------------------------------------------------------
// Kernel 3: main. One wave32 per batch row.
//  - Stage 2048-f32 context into LDS as f16 (per-wave 4KB region).
//  - Each lane owns A-row m = lane&15 (m=0:+center, 1..10:-negative, else 0).
//  - 64 x v_wmma_f32_16x16x32_f16 accumulate; C column 0 = 11 logits.
//  - softplus(-logit) partial per row -> scratch (deterministic reduce later).
// ---------------------------------------------------------------------------
__global__ __launch_bounds__(256) void necbow_main(
    const int*   __restrict__ windows,
    const int*   __restrict__ centers,
    const float* __restrict__ embedding,
    const float* __restrict__ output_embedding,
    const int*   __restrict__ noises,
    float*       __restrict__ partials,
    int B)
{
    __shared__ alignas(64) _Float16 ctx[8][CTX_DIM];   // 32 KB

    const int wave = threadIdx.x >> 5;
    const int lane = threadIdx.x & 31;
    const int b    = blockIdx.x * 8 + wave;
    if (b >= B) return;                                 // wave-uniform exit

    const int m    = lane & 15;                         // A-matrix row id
    const int half = lane >> 4;                         // lane group (K split)

    // ---- stage context (8 window embeddings) into LDS as f16 ----
    for (int w = 0; w < TWO_W; ++w) {
        const int row = windows[b * TWO_W + w];
        const float* src = embedding + (size_t)row * DIM;
        #pragma unroll
        for (int t = 0; t < DIM / 32; ++t)
            ctx[wave][w * DIM + t * 32 + lane] = (_Float16)src[t * 32 + lane];
    }

    // ---- per-lane output-embedding row + sign ----
    int   rowIdx = 0;
    float scale  = 0.f;
    if (m == 0)       { rowIdx = centers[b];                      scale =  1.f; }
    else if (m <= NS_CONST) { rowIdx = noises[b * NS_CONST + (m - 1)]; scale = -1.f; }
    const float* aptr = output_embedding + (size_t)rowIdx * CTX_DIM;

    v8f acc = {};
    for (int kk = 0; kk < KSTEPS; ++kk) {
        const int k0 = kk * 32 + half * 16;
        // A layout (16-bit, 16x32): this lane needs K = k0..k0+15 (contiguous, matches B)
        float4 f0 = *(const float4*)(aptr + k0);
        float4 f1 = *(const float4*)(aptr + k0 + 4);
        float4 f2 = *(const float4*)(aptr + k0 + 8);
        float4 f3 = *(const float4*)(aptr + k0 + 12);
        if (kk + 1 < KSTEPS)
            __builtin_prefetch(aptr + k0 + 32, 0, 1);   // global_prefetch_b8

        v16h A;
        A[0]  = (_Float16)(scale * f0.x); A[1]  = (_Float16)(scale * f0.y);
        A[2]  = (_Float16)(scale * f0.z); A[3]  = (_Float16)(scale * f0.w);
        A[4]  = (_Float16)(scale * f1.x); A[5]  = (_Float16)(scale * f1.y);
        A[6]  = (_Float16)(scale * f1.z); A[7]  = (_Float16)(scale * f1.w);
        A[8]  = (_Float16)(scale * f2.x); A[9]  = (_Float16)(scale * f2.y);
        A[10] = (_Float16)(scale * f2.z); A[11] = (_Float16)(scale * f2.w);
        A[12] = (_Float16)(scale * f3.x); A[13] = (_Float16)(scale * f3.y);
        A[14] = (_Float16)(scale * f3.z); A[15] = (_Float16)(scale * f3.w);

        // B layout (32x16): lanes 0-15 carry K=0..15, lanes 16-31 K=16..31;
        // all columns get the same context value -> contiguous 32B LDS read.
        v16h Bm = *(const v16h*)&ctx[wave][kk * 32 + half * 16];

        acc = __builtin_amdgcn_wmma_f32_16x16x32_f16(
            false, A, false, Bm, (short)0, acc, false, false);
    }

    // ---- column 0 of C: lane 0 holds M=0..7, lane 16 holds M=8..15 ----
    float part = 0.f;
    if (m == 0) {
        const int cnt = half ? (NS_CONST + 1 - 8) : 8;  // valid rows: 0..10
        for (int r = 0; r < cnt; ++r) {
            float x = acc[r];
            // softplus(-x), numerically stable
            part += fmaxf(-x, 0.f) + log1pf(expf(-fabsf(x)));
        }
    }
    float other = __shfl_down(part, 16, 32);            // lane16 -> lane0
    if (lane == 0) partials[b] = part + other;
}

// ---------------------------------------------------------------------------
// Kernel 4: deterministic single-block reduction of per-row partials.
// out[0] = loss (f32); out[1] = batch count (int32 bits, per reference tuple).
// ---------------------------------------------------------------------------
__global__ __launch_bounds__(256) void necbow_reduce(const float* __restrict__ partials,
                                                     int n, float* __restrict__ out,
                                                     int batch) {
    __shared__ float buf[256];
    float s = 0.f;
    for (int i = threadIdx.x; i < n; i += 256) s += partials[i];
    buf[threadIdx.x] = s;
    __syncthreads();
    for (int o = 128; o > 0; o >>= 1) {
        if (threadIdx.x < o) buf[threadIdx.x] += buf[threadIdx.x + o];
        __syncthreads();
    }
    if (threadIdx.x == 0) {
        out[0] = buf[0];
        ((int*)out)[1] = batch;
    }
}

// ---------------------------------------------------------------------------
extern "C" void kernel_launch(void* const* d_in, const int* in_sizes, int n_in,
                              void* d_out, int out_size, void* d_ws, size_t ws_size,
                              hipStream_t stream) {
    const int*   windows          = (const int*)d_in[0];
    const int*   centers          = (const int*)d_in[1];
    // d_in[2] = num_sampled (device scalar); NS hardcoded to 10 per reference
    const float* embedding        = (const float*)d_in[3];
    const float* output_embedding = (const float*)d_in[4];
    const float* weights          = (const float*)d_in[5];

    const int B = in_sizes[1];        // 8192
    const int V = in_sizes[5];        // 50000

    // workspace layout: [cdf: V f32][noises: B*NS i32][partials: B f32]
    size_t cdfBytes   = ((size_t)V * sizeof(float) + 255) & ~(size_t)255;
    size_t noiseBytes = ((size_t)B * NS_CONST * sizeof(int) + 255) & ~(size_t)255;
    float* cdf      = (float*)d_ws;
    int*   noises   = (int*)((char*)d_ws + cdfBytes);
    float* partials = (float*)((char*)d_ws + cdfBytes + noiseBytes);

    necbow_cdf_scan<<<1, 256, 0, stream>>>(weights, cdf, V);

    const int nSamp = B * NS_CONST;
    necbow_sample<<<(nSamp + 255) / 256, 256, 0, stream>>>(cdf, noises, nSamp, V);

    necbow_main<<<(B + 7) / 8, 256, 0, stream>>>(windows, centers, embedding,
                                                 output_embedding, noises,
                                                 partials, B);

    necbow_reduce<<<1, 256, 0, stream>>>(partials, B, (float*)d_out, B);
}
